// RotaryEmbeddingWithAttention_15891378995225
// MI455X (gfx1250) — compile-verified, matrix operand order checked by reading, not run on hardware
//
#include <hip/hip_runtime.h>
#include <hip/hip_bf16.h>

#define L_SEQ 4096
#define D_MODEL 1024
#define N_HEADS 16
#define N_WIN 8
#define W_WIN 512
#define HD 64

typedef __attribute__((ext_vector_type(16))) _Float16 v16h;
typedef __attribute__((ext_vector_type(8)))  float    v8f;

__device__ __forceinline__ v8f wmma_f16(v16h a, v16h b, v8f c) {
    // D = A(16x32 f16) * B(32x16 f16) + C(16x16 f32)
    return __builtin_amdgcn_wmma_f32_16x16x32_f16(
        /*neg_a=*/false, a, /*neg_b=*/false, b,
        /*c_mod=*/(short)0, c, /*reuse_a=*/false, /*reuse_b=*/false);
}

// DPP16 cross-lane reductions across each 16-lane row (wave32: lanes 0-15 / 16-31).
// Patterns: quad_perm xor1 (0xB1), quad_perm xor2 (0x4E),
//           row_half_mirror (0x141), row_mirror (0x140).
template <int CTRL>
__device__ __forceinline__ float dppf(float v) {
    int r = __builtin_amdgcn_update_dpp(0, __float_as_int(v), CTRL, 0xF, 0xF, true);
    return __int_as_float(r);
}
__device__ __forceinline__ float row16_max(float x) {
    x = fmaxf(x, dppf<0xB1>(x));
    x = fmaxf(x, dppf<0x4E>(x));
    x = fmaxf(x, dppf<0x141>(x));
    x = fmaxf(x, dppf<0x140>(x));
    return x;
}
__device__ __forceinline__ float row16_sum(float x) {
    x += dppf<0xB1>(x);
    x += dppf<0x4E>(x);
    x += dppf<0x141>(x);
    x += dppf<0x140>(x);
    return x;
}

// Build a 16x32 (f16) A/B operand fragment from a row-major [16, >=32] half tile.
// Per ISA: lanes 0-15 hold row M=lane (K-base 0), lanes 16-31 hold row M=lane-16
// (K-base 8). VGPR pair p holds K = {2p, 2p+1} for p<4, {16+2(p-4), ...} for p>=4.
__device__ __forceinline__ v16h load_frag_lds(const _Float16* base, int stride, int lane) {
    v16h f;
    const int r  = lane & 15;
    const int kb = (lane & 16) ? 8 : 0;
    const _Float16* row = base + r * stride;
#pragma unroll
    for (int p = 0; p < 8; ++p) {
        int k = ((p < 4) ? (2 * p) : (16 + 2 * (p - 4))) + kb;
        f[2 * p]     = row[k];
        f[2 * p + 1] = row[k + 1];
    }
    return f;
}

// Same fragment pattern, sourced from a global f32 row, scaled, converted to f16.
__device__ __forceinline__ v16h load_frag_gf32(const float* rowbase, int lane, int koff,
                                               float scale) {
    v16h f;
    const int kb = (lane & 16) ? 8 : 0;
#pragma unroll
    for (int p = 0; p < 8; ++p) {
        int k = ((p < 4) ? (2 * p) : (16 + 2 * (p - 4))) + kb + koff;
        f[2 * p]     = (_Float16)(rowbase[k] * scale);
        f[2 * p + 1] = (_Float16)(rowbase[k + 1] * scale);
    }
    return f;
}

// ---------------------------------------------------------------------------
// Kernel 1: rotary additive embedding
// ---------------------------------------------------------------------------
__global__ void __launch_bounds__(256) rotary_kernel(
    const float* __restrict__ x, const float* __restrict__ rf,
    float* __restrict__ x1)
{
    int idx = blockIdx.x * 256 + threadIdx.x;
    int l = idx >> 10;          // / D_MODEL
    int d = idx & 1023;
    int j = d & 511;
    float expo = (2.0f * (float)j) * (1.0f / 1024.0f);
    float inv  = __powf(10000.0f, expo);
    float ang  = rf[j] * (float)l / inv;
    float e    = (d < 512) ? __cosf(ang) : __sinf(ang);
    x1[idx] = x[idx] + e;
}

// ---------------------------------------------------------------------------
// Kernel 2: batched GEMM  C[M,N] = A[M,K] * B[N,K]^T + bias[N] (+ Res[M,N])
// Block: 256 threads (8 waves), 128x128 C tile, K-step 32.
// Register-staged pipeline + global_prefetch two K-steps ahead.
// ---------------------------------------------------------------------------
#define GTS 40   // LDS row stride (halves), padded to dodge bank conflicts

template <bool HAS_RES>
__global__ void __launch_bounds__(256) gemm_bias_kernel(
    const float* __restrict__ A, const float* __restrict__ Bw,
    const float* __restrict__ bias, const float* __restrict__ Res,
    float* __restrict__ C, int M, int N, int K,
    long long aStride, long long bStride, long long biasStride,
    long long resStride, long long cStride)
{
    __shared__ _Float16 As[128 * GTS];
    __shared__ _Float16 Bs[128 * GTS];

    const int z = blockIdx.z;
    A    += (size_t)z * aStride;
    Bw   += (size_t)z * bStride;
    bias += (size_t)z * biasStride;
    C    += (size_t)z * cStride;
    const float* R = HAS_RES ? (Res + (size_t)z * resStride) : nullptr;

    const int tid  = threadIdx.x;
    const int lane = tid & 31;
    const int wave = tid >> 5;
    const int m0 = (wave >> 1) * 32;   // 4 wave-rows of 32
    const int n0 = (wave & 1) * 64;    // 2 wave-cols of 64
    const int mBlock = blockIdx.y * 128;
    const int nBlock = blockIdx.x * 128;

    const v8f zero8 = {0.f, 0.f, 0.f, 0.f, 0.f, 0.f, 0.f, 0.f};
    v8f acc[2][4];
#pragma unroll
    for (int i = 0; i < 2; ++i)
#pragma unroll
        for (int j = 0; j < 4; ++j) acc[i][j] = zero8;

    const int ldRow = tid >> 1;        // 0..127
    const int ldCol = (tid & 1) * 16;  // 0 or 16
    const float* aBase = A + (size_t)(mBlock + ldRow) * K + ldCol;
    const float* bBase = Bw + (size_t)(nBlock + ldRow) * K + ldCol;

    // preload first K-tile into registers
    float4 ra[4], rb[4];
#pragma unroll
    for (int q = 0; q < 4; ++q) {
        ra[q] = ((const float4*)aBase)[q];
        rb[q] = ((const float4*)bBase)[q];
    }

    for (int k0 = 0; k0 < K; k0 += 32) {
        __syncthreads();
        // commit staged registers to LDS (f32 -> f16)
#pragma unroll
        for (int q = 0; q < 4; ++q) {
            _Float16* da = &As[ldRow * GTS + ldCol + q * 4];
            da[0] = (_Float16)ra[q].x; da[1] = (_Float16)ra[q].y;
            da[2] = (_Float16)ra[q].z; da[3] = (_Float16)ra[q].w;
            _Float16* db = &Bs[ldRow * GTS + ldCol + q * 4];
            db[0] = (_Float16)rb[q].x; db[1] = (_Float16)rb[q].y;
            db[2] = (_Float16)rb[q].z; db[3] = (_Float16)rb[q].w;
        }
        __syncthreads();

        // preload next K-tile while WMMAs below chew on LDS
        if (k0 + 32 < K) {
            const float4* ap4 = (const float4*)(aBase + k0 + 32);
            const float4* bp4 = (const float4*)(bBase + k0 + 32);
#pragma unroll
            for (int q = 0; q < 4; ++q) { ra[q] = ap4[q]; rb[q] = bp4[q]; }
            __builtin_prefetch(aBase + k0 + 64, 0, 3);   // global_prefetch_b8
            __builtin_prefetch(bBase + k0 + 64, 0, 3);
        }

        v16h af[2], bf[4];
#pragma unroll
        for (int i = 0; i < 2; ++i)
            af[i] = load_frag_lds(&As[(m0 + i * 16) * GTS], GTS, lane);
#pragma unroll
        for (int j = 0; j < 4; ++j)
            bf[j] = load_frag_lds(&Bs[(n0 + j * 16) * GTS], GTS, lane);
#pragma unroll
        for (int i = 0; i < 2; ++i)
#pragma unroll
            for (int j = 0; j < 4; ++j)
                acc[i][j] = wmma_f16(af[i], bf[j], acc[i][j]);
    }

    // C/D layout: VGPR r -> row = r + (lane>=16 ? 8 : 0), col = lane & 15
    const int rowOff = (lane & 16) ? 8 : 0;
    const int cn = lane & 15;
#pragma unroll
    for (int i = 0; i < 2; ++i) {
#pragma unroll
        for (int j = 0; j < 4; ++j) {
            const int gn = nBlock + n0 + j * 16 + cn;
            const float bv = bias[gn];
#pragma unroll
            for (int r = 0; r < 8; ++r) {
                const int gm = mBlock + m0 + i * 16 + rowOff + r;
                float v = acc[i][j][r] + bv;
                if (HAS_RES) v += R[(size_t)gm * N + gn];
                C[(size_t)gm * N + gn] = v;
            }
        }
    }
}

// ---------------------------------------------------------------------------
// Kernel 3: flash attention over qkv[L, 3*D] (f32), per (qblock, head, window).
// Block: 128 threads = 4 waves; wave owns 16 query rows; 64-row q-block.
// Key blocks of 64; online softmax with DPP16 row reductions (no LDS permutes);
// 1/sqrt(64) folded into the Q fragments.
// ---------------------------------------------------------------------------
#define KB   64
#define KS_S 72
#define VT_S 72
#define PB_S 68

__global__ void __launch_bounds__(128) attn_kernel(
    const float* __restrict__ qkv, float* __restrict__ out, int winRows)
{
    __shared__ _Float16 Ks[KB * KS_S];        // [key][d]   64 x 64
    __shared__ _Float16 Vt[HD * VT_S];        // [d][key]   64 x 64 (transposed)
    __shared__ _Float16 Pb[4 * 16 * PB_S];    // per-wave P relayout 16 x 64

    const int head = blockIdx.y;
    const int L0   = blockIdx.z * winRows;    // window base row
    const int tid  = threadIdx.x;
    const int lane = tid & 31;
    const int wave = tid >> 5;
    const int q0   = L0 + blockIdx.x * 64 + wave * 16;
    const int hb   = (lane & 16) ? 8 : 0;     // row-half offset in C/D layout
    const int cn   = lane & 15;
    const size_t RS = 3 * D_MODEL;

    // Q fragments (16 rows x 64 head-dim => two K=32 slices); fold in HD^-0.5
    const float* qrow = qkv + (size_t)(q0 + cn) * RS + head * HD;
    v16h qf0 = load_frag_gf32(qrow, lane, 0, 0.125f);
    v16h qf1 = load_frag_gf32(qrow, lane, 32, 0.125f);

    const v8f zero8 = {0.f, 0.f, 0.f, 0.f, 0.f, 0.f, 0.f, 0.f};
    v8f o[4];
    float mrun[8], lrun[8];
#pragma unroll
    for (int t = 0; t < 4; ++t) o[t] = zero8;
#pragma unroll
    for (int r = 0; r < 8; ++r) { mrun[r] = -3.0e38f; lrun[r] = 0.0f; }

    const int key = tid >> 1;          // 0..63
    const int db  = (tid & 1) * 32;    // 0 or 32

    for (int kb0 = 0; kb0 < winRows; kb0 += KB) {
        __syncthreads();
        // stage K block [64 x 64] and V block transposed [64 x 64]
        {
            const float* krow = qkv + (size_t)(L0 + kb0 + key) * RS + D_MODEL + head * HD + db;
            const float4* kp4 = (const float4*)krow;
            const float4* vp4 = (const float4*)(krow + D_MODEL);
#pragma unroll
            for (int q = 0; q < 8; ++q) {
                float4 kv = kp4[q];
                _Float16* dk = &Ks[key * KS_S + db + q * 4];
                dk[0] = (_Float16)kv.x; dk[1] = (_Float16)kv.y;
                dk[2] = (_Float16)kv.z; dk[3] = (_Float16)kv.w;
                float4 vv = vp4[q];
                Vt[(db + q * 4 + 0) * VT_S + key] = (_Float16)vv.x;
                Vt[(db + q * 4 + 1) * VT_S + key] = (_Float16)vv.y;
                Vt[(db + q * 4 + 2) * VT_S + key] = (_Float16)vv.z;
                Vt[(db + q * 4 + 3) * VT_S + key] = (_Float16)vv.w;
            }
            if (kb0 + KB < winRows) {
                __builtin_prefetch(krow + KB * RS, 0, 3);            // next K block
                __builtin_prefetch(krow + KB * RS + D_MODEL, 0, 3);  // next V block
            }
        }
        __syncthreads();

        // S = Q * K^T  (four 16-key subtiles, head-dim reduced in 2 x K=32)
        v8f s[4];
#pragma unroll
        for (int ks = 0; ks < 4; ++ks) {
            s[ks] = wmma_f16(qf0, load_frag_lds(&Ks[ks * 16 * KS_S],      KS_S, lane), zero8);
            s[ks] = wmma_f16(qf1, load_frag_lds(&Ks[ks * 16 * KS_S + 32], KS_S, lane), s[ks]);
        }

        // online softmax; row reductions via DPP16 (stay within each 16-lane half)
        _Float16* pb = &Pb[wave * 16 * PB_S];
#pragma unroll
        for (int r = 0; r < 8; ++r) {
            float v0 = s[0][r], v1 = s[1][r], v2 = s[2][r], v3 = s[3][r];
            float mx = row16_max(fmaxf(fmaxf(v0, v1), fmaxf(v2, v3)));
            float mnew  = fmaxf(mrun[r], mx);
            float alpha = __expf(mrun[r] - mnew);
            mrun[r] = mnew;
            float p0 = __expf(v0 - mnew);
            float p1 = __expf(v1 - mnew);
            float p2 = __expf(v2 - mnew);
            float p3 = __expf(v3 - mnew);
            float ps = row16_sum((p0 + p1) + (p2 + p3));
            lrun[r] = lrun[r] * alpha + ps;
            o[0][r] *= alpha; o[1][r] *= alpha; o[2][r] *= alpha; o[3][r] *= alpha;
            const int prow = r + hb;
            pb[prow * PB_S + cn]      = (_Float16)p0;
            pb[prow * PB_S + 16 + cn] = (_Float16)p1;
            pb[prow * PB_S + 32 + cn] = (_Float16)p2;
            pb[prow * PB_S + 48 + cn] = (_Float16)p3;
        }

        // O += P[16x64] * V[64x64]  (4 d-slabs x 2 key-halves; B-operand from Vt)
        v16h pf0 = load_frag_lds(pb,      PB_S, lane);
        v16h pf1 = load_frag_lds(pb + 32, PB_S, lane);
#pragma unroll
        for (int t = 0; t < 4; ++t) {
            o[t] = wmma_f16(pf0, load_frag_lds(&Vt[(t * 16) * VT_S],      VT_S, lane), o[t]);
            o[t] = wmma_f16(pf1, load_frag_lds(&Vt[(t * 16) * VT_S + 32], VT_S, lane), o[t]);
        }
    }

    // normalize and store: out[L, D] f32
#pragma unroll
    for (int r = 0; r < 8; ++r) {
        const float invl = 1.0f / lrun[r];
        const int gm = q0 + hb + r;
#pragma unroll
        for (int t = 0; t < 4; ++t) {
            out[(size_t)gm * D_MODEL + head * HD + t * 16 + cn] = o[t][r] * invl;
        }
    }
}

// ---------------------------------------------------------------------------
extern "C" void kernel_launch(void* const* d_in, const int* in_sizes, int n_in,
                              void* d_out, int out_size, void* d_ws, size_t ws_size,
                              hipStream_t stream) {
    (void)in_sizes; (void)n_in; (void)out_size; (void)ws_size;
    const float* x   = (const float*)d_in[0];
    const float* rf  = (const float*)d_in[1];
    const float* wwi = (const float*)d_in[2];  // [8, 3072, 1024]
    const float* wbi = (const float*)d_in[3];  // [8, 3072]
    const float* wwo = (const float*)d_in[4];  // [8, 1024, 1024]
    const float* wbo = (const float*)d_in[5];  // [8, 1024]
    const float* fwi = (const float*)d_in[6];  // [3072, 1024]
    const float* fbi = (const float*)d_in[7];  // [3072]
    const float* fwo = (const float*)d_in[8];  // [1024, 1024]
    const float* fbo = (const float*)d_in[9];  // [1024]
    float* out = (float*)d_out;

    float* ws   = (float*)d_ws;
    float* x1   = ws;                                       // [4096,1024]
    float* qkv  = x1  + (size_t)L_SEQ * D_MODEL;            // [4096,3072]
    float* attn = qkv + (size_t)L_SEQ * 3 * D_MODEL;        // [4096,1024]
    float* x2   = attn + (size_t)L_SEQ * D_MODEL;           // [4096,1024]

    // 1) rotary additive embedding
    rotary_kernel<<<(L_SEQ * D_MODEL) / 256, 256, 0, stream>>>(x, rf, x1);

    // 2) windowed QKV: per window, [512,1024] x [3072,1024]^T + b -> [512,3072]
    gemm_bias_kernel<false><<<dim3(3072 / 128, W_WIN / 128, N_WIN), 256, 0, stream>>>(
        x1, wwi, wbi, nullptr, qkv, W_WIN, 3 * D_MODEL, D_MODEL,
        (long long)W_WIN * D_MODEL, (long long)3 * D_MODEL * D_MODEL,
        (long long)3 * D_MODEL, 0, (long long)W_WIN * 3 * D_MODEL);

    // 3) windowed attention (8 qblocks x 16 heads x 8 windows)
    attn_kernel<<<dim3(W_WIN / 64, N_HEADS, N_WIN), 128, 0, stream>>>(qkv, attn, W_WIN);

    // 4) windowed out-proj + residual: x2 = x1 + attn @ w_out^T + b_out
    gemm_bias_kernel<true><<<dim3(D_MODEL / 128, W_WIN / 128, N_WIN), 256, 0, stream>>>(
        attn, wwo, wbo, x1, x2, W_WIN, D_MODEL, D_MODEL,
        (long long)W_WIN * D_MODEL, (long long)D_MODEL * D_MODEL,
        (long long)D_MODEL, (long long)W_WIN * D_MODEL, (long long)W_WIN * D_MODEL);

    // 5) final QKV: [4096,1024] x [3072,1024]^T + b -> [4096,3072]
    gemm_bias_kernel<false><<<dim3(3072 / 128, L_SEQ / 128, 1), 256, 0, stream>>>(
        x2, fwi, fbi, nullptr, qkv, L_SEQ, 3 * D_MODEL, D_MODEL, 0, 0, 0, 0, 0);

    // 6) global attention
    attn_kernel<<<dim3(L_SEQ / 64, N_HEADS, 1), 128, 0, stream>>>(qkv, attn, L_SEQ);

    // 7) final out-proj -> d_out
    gemm_bias_kernel<false><<<dim3(D_MODEL / 128, L_SEQ / 128, 1), 256, 0, stream>>>(
        attn, fwo, fbo, nullptr, out, L_SEQ, D_MODEL, D_MODEL, 0, 0, 0, 0, 0);
}
